// RWKVTimeMixing_22574348108495
// MI455X (gfx1250) — compile-verified, matrix operand order checked by reading, not run on hardware
//
#include <hip/hip_runtime.h>
#include <hip/hip_bf16.h>
#include <math.h>

#define BB   4
#define TT   2048
#define CCH  2048
#define RR   256
#define MTOT (BB * TT)          // 8192 token rows

typedef __attribute__((ext_vector_type(16))) __bf16       bf16x16;
typedef __attribute__((ext_vector_type(8)))  __bf16       bf16x8;
typedef __attribute__((ext_vector_type(2)))  __bf16       bf16x2;
typedef __attribute__((ext_vector_type(8)))  float        f32x8;
typedef __attribute__((ext_vector_type(4)))  unsigned int u32x4;

union Frag { bf16x16 v; u32x4 q[2]; };
union V8   { u32x4 q; bf16x8 v; };
union U32B { unsigned u; bf16x2 v; };

static __device__ __forceinline__ f32x8 wmma_bf16(Frag a, Frag b, f32x8 c) {
    // D = A(16x32 bf16) * B(32x16 bf16) + C(16x16 f32)
    return __builtin_amdgcn_wmma_f32_16x16x32_bf16(
        /*neg_a=*/false, a.v, /*neg_b=*/false, b.v,
        /*c_mod=*/(short)0, c, /*reuse_a=*/false, /*reuse_b=*/false);
}

// Async global -> LDS copy, 16B per lane.  VDST = LDS byte address (VGPR),
// VADDR = 64-bit global address (VGPR pair), GV mode.  Tracked by ASYNCcnt.
static __device__ __forceinline__ void async_g2l_b128(const void* g, void* lds) {
    unsigned lds_off = (unsigned)(size_t)lds;   // flat addr[31:0] == LDS offset
    asm volatile("global_load_async_to_lds_b128 %0, %1, off"
                 :: "v"(lds_off), "v"(g) : "memory");
}
static __device__ __forceinline__ void wait_async0() {
#if __has_builtin(__builtin_amdgcn_s_wait_asynccnt)
    __builtin_amdgcn_s_wait_asynccnt(0);
#else
    asm volatile("s_wait_asynccnt 0x0" ::: "memory");
#endif
}

// ---------------------------------------------------------------- converts
__global__ void f32_to_bf16_kernel(const float* __restrict__ src,
                                   __bf16* __restrict__ dst, int n) {
    int i = blockIdx.x * 256 + threadIdx.x;
    if (i < n) dst[i] = (__bf16)src[i];
}

// src is K x N row-major fp32; dst is N x K row-major bf16 (transposed)
__global__ void transpose_cvt_kernel(const float* __restrict__ src,
                                     __bf16* __restrict__ dst, int K, int N) {
    int idx = blockIdx.x * 256 + threadIdx.x;
    if (idx < K * N) {
        int k = idx / N, n = idx - k * N;
        dst[(size_t)n * K + k] = (__bf16)src[idx];
    }
}

// ---------------------------------------------------------------- layernorm
__global__ __launch_bounds__(256) void ln_kernel(const float* __restrict__ x,
                                                 const float* __restrict__ gamma,
                                                 const float* __restrict__ beta,
                                                 __bf16* __restrict__ xn) {
    __shared__ float ssum[256];
    __shared__ float ssq[256];
    const int row = blockIdx.x;                 // 0 .. MTOT-1
    const float* xr = x + (size_t)row * CCH;
    float s = 0.f, s2 = 0.f;
    for (int j = threadIdx.x; j < CCH; j += 256) {
        float v = xr[j]; s += v; s2 += v * v;
    }
    ssum[threadIdx.x] = s; ssq[threadIdx.x] = s2;
    __syncthreads();
    for (int off = 128; off > 0; off >>= 1) {
        if (threadIdx.x < off) {
            ssum[threadIdx.x] += ssum[threadIdx.x + off];
            ssq[threadIdx.x]  += ssq[threadIdx.x + off];
        }
        __syncthreads();
    }
    const float mu  = ssum[0] * (1.0f / CCH);
    const float var = ssq[0] * (1.0f / CCH) - mu * mu;
    const float rs  = rsqrtf(var + 1e-5f);
    for (int j = threadIdx.x; j < CCH; j += 256) {
        float v = (xr[j] - mu) * rs * gamma[j] + beta[j];
        xn[(size_t)row * CCH + j] = (__bf16)v;
    }
}

// ---------------------------------------------------------------- GEMM stage 1
// qr/kr/vr = mix(8192x2048) @ {wq_a,wk_a,wv_a}; token-shift fused into staging
// using packed bf16 math; weights pre-transposed N-major (ld=K).
// A tile: manual stage (mix math).  B tiles: async global->LDS.
__global__ __launch_bounds__(256) void gemm_lowrank_down(
    const __bf16* __restrict__ XN,
    const __bf16* __restrict__ B0, const __bf16* __restrict__ B1,
    const __bf16* __restrict__ B2,
    __bf16* __restrict__ O0, __bf16* __restrict__ O1, __bf16* __restrict__ O2) {
    constexpr int BM = 64, BN = 64, BK = 64, K = CCH, N = RR;
    constexpr int NCH = K / BK;
    __shared__ __align__(16) __bf16 As[2][BM * BK];
    __shared__ __align__(16) __bf16 Bs[2][3][BN * BK];
    const int m0 = blockIdx.y * BM, n0 = blockIdx.x * BN;
    const int tid = threadIdx.x, lane = tid & 31, wave = tid >> 5;
    const int wm = wave >> 1, wn = wave & 1;
    const __bf16* Bp[3] = {B0, B1, B2};

    const bf16x8 halfv = {(__bf16)0.5f, (__bf16)0.5f, (__bf16)0.5f, (__bf16)0.5f,
                          (__bf16)0.5f, (__bf16)0.5f, (__bf16)0.5f, (__bf16)0.5f};

    // loop-invariant staging addressing: 2 passes of 8-elem vectors
    const int srow = tid >> 3, scol = (tid & 7) * 8;
    const __bf16* acur[2];  const __bf16* aprv[2];  u32x4 pmask[2];
    const __bf16* bbase[3][2];
    #pragma unroll
    for (int p = 0; p < 2; ++p) {
        size_t r = (size_t)(m0 + srow + 32 * p);
        const bool t0 = ((r & (TT - 1)) == 0);
        acur[p]  = XN + r * K + scol;
        aprv[p]  = t0 ? acur[p] : (acur[p] - K);       // clamp at row 0
        const unsigned mk = t0 ? 0u : 0xffffffffu;     // zero prev at t==0
        pmask[p] = (u32x4){mk, mk, mk, mk};
        #pragma unroll
        for (int h = 0; h < 3; ++h)
            bbase[h][p] = Bp[h] + (size_t)(n0 + srow + 32 * p) * K + scol;
    }

    f32x8 acc[3][2] = {};
    u32x4 nAc[2], nAp[2];

    // ---- chunk 0: async B, manual (mix) A
    #pragma unroll
    for (int h = 0; h < 3; ++h)
        #pragma unroll
        for (int p = 0; p < 2; ++p)
            async_g2l_b128(bbase[h][p], &Bs[0][h][(srow + 32 * p) * BK + scol]);
    #pragma unroll
    for (int p = 0; p < 2; ++p) { nAc[p] = *(const u32x4*)acur[p];
                                  nAp[p] = *(const u32x4*)aprv[p]; }
    #pragma unroll
    for (int p = 0; p < 2; ++p) {
        V8 a, b, m; a.q = nAc[p]; b.q = nAp[p] & pmask[p];
        m.v = (a.v + b.v) * halfv;                     // packed bf16 mix
        *(u32x4*)&As[0][(srow + 32 * p) * BK + scol] = m.q;
    }
    wait_async0();
    __syncthreads();

    const int arow = wm * 16 + (lane & 15);
    const int kg   = (lane >> 4) * 16;
    for (int c = 0; c < NCH; ++c) {
        const int buf = c & 1;
        const bool more = (c + 1 < NCH);
        if (more) {                       // launch next chunk: async B + A regs
            const int k1 = (c + 1) * BK;
            const int nb = buf ^ 1;
            #pragma unroll
            for (int h = 0; h < 3; ++h)
                #pragma unroll
                for (int p = 0; p < 2; ++p)
                    async_g2l_b128(bbase[h][p] + k1,
                                   &Bs[nb][h][(srow + 32 * p) * BK + scol]);
            #pragma unroll
            for (int p = 0; p < 2; ++p) { nAc[p] = *(const u32x4*)(acur[p] + k1);
                                          nAp[p] = *(const u32x4*)(aprv[p] + k1); }
        }
        // compute current chunk: batch all LDS fragment loads, then wmma chain
        #pragma unroll
        for (int ks = 0; ks < 2; ++ks) {
            Frag a, bf_[3][2];
            a.q[0] = *(const u32x4*)&As[buf][arow * BK + ks * 32 + kg];
            a.q[1] = *(const u32x4*)&As[buf][arow * BK + ks * 32 + kg + 8];
            #pragma unroll
            for (int h = 0; h < 3; ++h)
                #pragma unroll
                for (int cn = 0; cn < 2; ++cn) {
                    const int bcol = wn * 32 + cn * 16 + (lane & 15);
                    bf_[h][cn].q[0] =
                        *(const u32x4*)&Bs[buf][h][bcol * BK + ks * 32 + kg];
                    bf_[h][cn].q[1] =
                        *(const u32x4*)&Bs[buf][h][bcol * BK + ks * 32 + kg + 8];
                }
            #pragma unroll
            for (int h = 0; h < 3; ++h)
                #pragma unroll
                for (int cn = 0; cn < 2; ++cn)
                    acc[h][cn] = wmma_bf16(a, bf_[h][cn], acc[h][cn]);
        }
        if (more) {                       // finish staging the next chunk
            const int nb = buf ^ 1;
            #pragma unroll
            for (int p = 0; p < 2; ++p) {
                V8 a, b, m; a.q = nAc[p]; b.q = nAp[p] & pmask[p];
                m.v = (a.v + b.v) * halfv;
                *(u32x4*)&As[nb][(srow + 32 * p) * BK + scol] = m.q;
            }
            wait_async0();
        }
        __syncthreads();
    }

    __bf16* Op[3] = {O0, O1, O2};
    const int mbase = m0 + wm * 16 + (lane >> 4) * 8;
    #pragma unroll
    for (int h = 0; h < 3; ++h)
        #pragma unroll
        for (int cn = 0; cn < 2; ++cn) {
            int ncol = n0 + wn * 32 + cn * 16 + (lane & 15);
            #pragma unroll
            for (int i = 0; i < 8; ++i)
                Op[h][(size_t)(mbase + i) * N + ncol] = (__bf16)acc[h][cn][i];
        }
}

// ---------------------------------------------------------------- GEMM stage 2
// sq = sigmoid(qr@wq_b), kv = (kr@wk_b)*(vr@wv_b); all staging async g->LDS
__global__ __launch_bounds__(256) void gemm_lowrank_up(
    const __bf16* __restrict__ Aq, const __bf16* __restrict__ Ak,
    const __bf16* __restrict__ Av,
    const __bf16* __restrict__ Bq, const __bf16* __restrict__ Bk,
    const __bf16* __restrict__ Bv,
    __bf16* __restrict__ sq, __bf16* __restrict__ kv) {
    constexpr int BM = 64, BN = 64, BK = 64, K = RR, N = CCH;
    constexpr int NCH = K / BK;
    __shared__ __align__(16) __bf16 As[2][3][BM * BK];
    __shared__ __align__(16) __bf16 Bs[2][3][BN * BK];
    const int m0 = blockIdx.y * BM, n0 = blockIdx.x * BN;
    const int tid = threadIdx.x, lane = tid & 31, wave = tid >> 5;
    const int wm = wave >> 1, wn = wave & 1;
    const __bf16* Ap[3] = {Aq, Ak, Av};
    const __bf16* Bp[3] = {Bq, Bk, Bv};

    const int srow = tid >> 3, scol = (tid & 7) * 8;
    const __bf16* abase[3][2];  const __bf16* bbase[3][2];
    #pragma unroll
    for (int h = 0; h < 3; ++h)
        #pragma unroll
        for (int p = 0; p < 2; ++p) {
            abase[h][p] = Ap[h] + (size_t)(m0 + srow + 32 * p) * K + scol;
            bbase[h][p] = Bp[h] + (size_t)(n0 + srow + 32 * p) * K + scol;
        }

    f32x8 acc[3][2] = {};
    #pragma unroll
    for (int h = 0; h < 3; ++h)
        #pragma unroll
        for (int p = 0; p < 2; ++p) {
            async_g2l_b128(abase[h][p], &As[0][h][(srow + 32 * p) * BK + scol]);
            async_g2l_b128(bbase[h][p], &Bs[0][h][(srow + 32 * p) * BK + scol]);
        }
    wait_async0();
    __syncthreads();

    const int arow = wm * 16 + (lane & 15);
    const int kg   = (lane >> 4) * 16;
    for (int c = 0; c < NCH; ++c) {
        const int buf = c & 1;
        const bool more = (c + 1 < NCH);
        if (more) {
            const int k1 = (c + 1) * BK;
            const int nb = buf ^ 1;
            #pragma unroll
            for (int h = 0; h < 3; ++h)
                #pragma unroll
                for (int p = 0; p < 2; ++p) {
                    async_g2l_b128(abase[h][p] + k1,
                                   &As[nb][h][(srow + 32 * p) * BK + scol]);
                    async_g2l_b128(bbase[h][p] + k1,
                                   &Bs[nb][h][(srow + 32 * p) * BK + scol]);
                }
        }
        #pragma unroll
        for (int ks = 0; ks < 2; ++ks) {
            Frag afr[3];
            #pragma unroll
            for (int h = 0; h < 3; ++h) {
                afr[h].q[0] = *(const u32x4*)&As[buf][h][arow * BK + ks * 32 + kg];
                afr[h].q[1] = *(const u32x4*)&As[buf][h][arow * BK + ks * 32 + kg + 8];
            }
            #pragma unroll
            for (int h = 0; h < 3; ++h) {
                Frag b0, b1;
                const int bc0 = (wn * 32 + (lane & 15)) * BK + ks * 32 + kg;
                const int bc1 = (wn * 32 + 16 + (lane & 15)) * BK + ks * 32 + kg;
                b0.q[0] = *(const u32x4*)&Bs[buf][h][bc0];
                b0.q[1] = *(const u32x4*)&Bs[buf][h][bc0 + 8];
                b1.q[0] = *(const u32x4*)&Bs[buf][h][bc1];
                b1.q[1] = *(const u32x4*)&Bs[buf][h][bc1 + 8];
                acc[h][0] = wmma_bf16(afr[h], b0, acc[h][0]);
                acc[h][1] = wmma_bf16(afr[h], b1, acc[h][1]);
            }
        }
        if (more) wait_async0();
        __syncthreads();
    }

    const int mbase = m0 + wm * 16 + (lane >> 4) * 8;
    #pragma unroll
    for (int cn = 0; cn < 2; ++cn) {
        int ncol = n0 + wn * 32 + cn * 16 + (lane & 15);
        #pragma unroll
        for (int i = 0; i < 8; ++i) {
            size_t p = (size_t)(mbase + i) * N + ncol;
            float qv = acc[0][cn][i];
            sq[p] = (__bf16)(1.0f / (1.0f + __expf(-qv)));
            kv[p] = (__bf16)(acc[1][cn][i] * acc[2][cn][i]);
        }
    }
}

// ---------------------------------------------------------------- recurrence
// Two channels per thread: u32 packed bf16 loads/stores, two independent
// h-chains per thread for ILP against the serial-FMA dependency.
__global__ __launch_bounds__(256) void scan_kernel(
    const unsigned* __restrict__ sq2, const unsigned* __restrict__ kv2,
    const float* __restrict__ h_state, const float* __restrict__ time_decay,
    unsigned* __restrict__ y2, float* __restrict__ h_out) {
    const int idx2 = blockIdx.x * 256 + threadIdx.x;   // 0 .. B*C/2-1
    const int b = idx2 / (CCH / 2);
    const int c = (idx2 % (CCH / 2)) * 2;
    const float d0 = __expf(-__expf(time_decay[c]));
    const float d1 = __expf(-__expf(time_decay[c + 1]));
    float h0 = h_state[b * CCH + c];
    float h1 = h_state[b * CCH + c + 1];
    const size_t base = ((size_t)b * TT * CCH + c) >> 1;   // in u32 units
    #pragma unroll 4
    for (int t = 0; t < TT; ++t) {
        size_t p = base + (size_t)t * (CCH / 2);
        U32B kvp, sqp, yp;
        kvp.u = kv2[p]; sqp.u = sq2[p];
        h0 = d0 * h0 + (float)kvp.v[0];
        h1 = d1 * h1 + (float)kvp.v[1];
        yp.v[0] = (__bf16)((float)sqp.v[0] * h0);
        yp.v[1] = (__bf16)((float)sqp.v[1] * h1);
        y2[p] = yp.u;
    }
    h_out[b * CCH + c]     = h0;
    h_out[b * CCH + c + 1] = h1;
}

// ---------------------------------------------------------------- output GEMM
// out = y(8192x2048) @ out_w^T + out_b ; all staging async g->LDS
__global__ __launch_bounds__(256) void gemm_out_kernel(
    const __bf16* __restrict__ A, const __bf16* __restrict__ W,
    const float* __restrict__ bias, float* __restrict__ out) {
    constexpr int BM = 64, BN = 64, BK = 64, K = CCH, N = CCH;
    constexpr int NCH = K / BK;
    __shared__ __align__(16) __bf16 As[2][BM * BK];
    __shared__ __align__(16) __bf16 Bs[2][BN * BK];
    const int m0 = blockIdx.y * BM, n0 = blockIdx.x * BN;
    const int tid = threadIdx.x, lane = tid & 31, wave = tid >> 5;
    const int wm = wave >> 1, wn = wave & 1;

    const int srow = tid >> 3, scol = (tid & 7) * 8;
    const __bf16* abase[2];  const __bf16* bbase[2];
    #pragma unroll
    for (int p = 0; p < 2; ++p) {
        abase[p] = A + (size_t)(m0 + srow + 32 * p) * K + scol;
        bbase[p] = W + (size_t)(n0 + srow + 32 * p) * K + scol;
    }

    f32x8 acc[2] = {};
    #pragma unroll
    for (int p = 0; p < 2; ++p) {
        async_g2l_b128(abase[p], &As[0][(srow + 32 * p) * BK + scol]);
        async_g2l_b128(bbase[p], &Bs[0][(srow + 32 * p) * BK + scol]);
    }
    wait_async0();
    __syncthreads();

    const int arow = wm * 16 + (lane & 15);
    const int kg   = (lane >> 4) * 16;
    for (int c = 0; c < NCH; ++c) {
        const int buf = c & 1;
        const bool more = (c + 1 < NCH);
        if (more) {
            const int k1 = (c + 1) * BK;
            const int nb = buf ^ 1;
            #pragma unroll
            for (int p = 0; p < 2; ++p) {
                async_g2l_b128(abase[p] + k1, &As[nb][(srow + 32 * p) * BK + scol]);
                async_g2l_b128(bbase[p] + k1, &Bs[nb][(srow + 32 * p) * BK + scol]);
            }
        }
        #pragma unroll
        for (int ks = 0; ks < 2; ++ks) {
            Frag a, b0, b1;
            a.q[0] = *(const u32x4*)&As[buf][arow * BK + ks * 32 + kg];
            a.q[1] = *(const u32x4*)&As[buf][arow * BK + ks * 32 + kg + 8];
            const int bc0 = (wn * 32 + (lane & 15)) * BK + ks * 32 + kg;
            const int bc1 = (wn * 32 + 16 + (lane & 15)) * BK + ks * 32 + kg;
            b0.q[0] = *(const u32x4*)&Bs[buf][bc0];
            b0.q[1] = *(const u32x4*)&Bs[buf][bc0 + 8];
            b1.q[0] = *(const u32x4*)&Bs[buf][bc1];
            b1.q[1] = *(const u32x4*)&Bs[buf][bc1 + 8];
            acc[0] = wmma_bf16(a, b0, acc[0]);
            acc[1] = wmma_bf16(a, b1, acc[1]);
        }
        if (more) wait_async0();
        __syncthreads();
    }

    const int mbase = m0 + wm * 16 + (lane >> 4) * 8;
    #pragma unroll
    for (int cn = 0; cn < 2; ++cn) {
        int ncol = n0 + wn * 32 + cn * 16 + (lane & 15);
        float bv = bias[ncol];
        #pragma unroll
        for (int i = 0; i < 8; ++i)
            out[(size_t)(mbase + i) * N + ncol] = acc[cn][i] + bv;
    }
}

// ---------------------------------------------------------------- launch
extern "C" void kernel_launch(void* const* d_in, const int* in_sizes, int n_in,
                              void* d_out, int out_size, void* d_ws, size_t ws_size,
                              hipStream_t stream) {
    (void)in_sizes; (void)n_in; (void)out_size; (void)ws_size;
    const float* x          = (const float*)d_in[0];
    const float* h_state    = (const float*)d_in[1];
    const float* ln_gamma   = (const float*)d_in[2];
    const float* ln_beta    = (const float*)d_in[3];
    const float* wq_a       = (const float*)d_in[4];
    const float* wq_b       = (const float*)d_in[5];
    const float* wk_a       = (const float*)d_in[6];
    const float* wk_b       = (const float*)d_in[7];
    const float* wv_a       = (const float*)d_in[8];
    const float* wv_b       = (const float*)d_in[9];
    const float* out_w      = (const float*)d_in[10];
    const float* out_b      = (const float*)d_in[11];
    const float* time_decay = (const float*)d_in[12];

    char* ws = (char*)d_ws;
    size_t off = 0;
    auto take = [&](size_t bytes) -> char* {
        char* p = ws + off;
        off = (off + bytes + 255) & ~(size_t)255;
        return p;
    };
    const size_t NTOK = (size_t)MTOT * CCH;            // 16.7M elems
    const size_t NLOW = (size_t)CCH * RR;              // 512K elems
    __bf16* xn     = (__bf16*)take(NTOK * 2);
    __bf16* wqa_t  = (__bf16*)take(NLOW * 2);          // 256 x 2048 (N-major)
    __bf16* wka_t  = (__bf16*)take(NLOW * 2);
    __bf16* wva_t  = (__bf16*)take(NLOW * 2);
    __bf16* wqb_t  = (__bf16*)take(NLOW * 2);          // 2048 x 256 (N-major)
    __bf16* wkb_t  = (__bf16*)take(NLOW * 2);
    __bf16* wvb_t  = (__bf16*)take(NLOW * 2);
    __bf16* outw_h = (__bf16*)take((size_t)CCH * CCH * 2);
    __bf16* qr     = (__bf16*)take((size_t)MTOT * RR * 2);
    __bf16* kr     = (__bf16*)take((size_t)MTOT * RR * 2);
    __bf16* vr     = (__bf16*)take((size_t)MTOT * RR * 2);
    __bf16* sqb    = (__bf16*)take(NTOK * 2);
    __bf16* kvb    = (__bf16*)take(NTOK * 2);
    __bf16* yb     = (__bf16*)take(NTOK * 2);

    // 1. weight conversion (+ transpose to N-major for the low-rank mats)
    const int nlow_blocks = (int)((NLOW + 255) / 256);
    transpose_cvt_kernel<<<nlow_blocks, 256, 0, stream>>>(wq_a, wqa_t, CCH, RR);
    transpose_cvt_kernel<<<nlow_blocks, 256, 0, stream>>>(wk_a, wka_t, CCH, RR);
    transpose_cvt_kernel<<<nlow_blocks, 256, 0, stream>>>(wv_a, wva_t, CCH, RR);
    transpose_cvt_kernel<<<nlow_blocks, 256, 0, stream>>>(wq_b, wqb_t, RR, CCH);
    transpose_cvt_kernel<<<nlow_blocks, 256, 0, stream>>>(wk_b, wkb_t, RR, CCH);
    transpose_cvt_kernel<<<nlow_blocks, 256, 0, stream>>>(wv_b, wvb_t, RR, CCH);
    f32_to_bf16_kernel<<<(CCH * CCH + 255) / 256, 256, 0, stream>>>(out_w, outw_h,
                                                                    CCH * CCH);

    // 2. layernorm -> xn (bf16); token shift is fused into stage-1 staging
    ln_kernel<<<MTOT, 256, 0, stream>>>(x, ln_gamma, ln_beta, xn);

    // 3. fused low-rank down-projection (q,k,v share A traffic)
    dim3 g1(RR / 64, MTOT / 64);
    gemm_lowrank_down<<<g1, 256, 0, stream>>>(xn, wqa_t, wka_t, wva_t, qr, kr, vr);

    // 4. fused up-projection with sigmoid / k*v epilogue
    dim3 g2(CCH / 64, MTOT / 64);
    gemm_lowrank_up<<<g2, 256, 0, stream>>>(qr, kr, vr, wqb_t, wkb_t, wvb_t, sqb, kvb);

    // 5. time-decay recurrence (2 channels/thread); h_final to tail of d_out
    float* out_f = (float*)d_out;
    scan_kernel<<<(BB * CCH / 2) / 256, 256, 0, stream>>>(
        (const unsigned*)sqb, (const unsigned*)kvb, h_state, time_decay,
        (unsigned*)yb, out_f + NTOK);

    // 6. output projection + bias
    gemm_out_kernel<<<g2, 256, 0, stream>>>(yb, outw_h, out_b, out_f);
}
